// WindowedAttention_38225208934920
// MI455X (gfx1250) — matrix operand order repaired and checked by reading, not compiled, not run on hardware
//
#include <hip/hip_runtime.h>
#include <hip/hip_bf16.h>

typedef __bf16 bf16_t;
typedef __attribute__((ext_vector_type(16))) __bf16 v16bf;
typedef __attribute__((ext_vector_type(8)))  __bf16 v8bf;
typedef __attribute__((ext_vector_type(4)))  __bf16 v4bf;
typedef __attribute__((ext_vector_type(8)))  float  v8f;

#define T_SEQ   4096
#define DMODEL  1024
#define EQKV    3072
#define NHEAD   16
#define DHEAD   64
#define WIN     512
#define NPERS   16

// gfx1250 async global->LDS copies (ASYNCcnt-tracked). Set to 0 if the
// assembler rejects the mnemonic.
#define USE_ASYNC 1

__device__ __forceinline__ v16bf cat16(v8bf lo, v8bf hi) {
  return __builtin_shufflevector(lo, hi, 0,1,2,3,4,5,6,7,8,9,10,11,12,13,14,15);
}

__device__ __forceinline__ bool mask_ok(int q, int k) {
  if (k < NPERS) return true;
  return (q >= NPERS) && (k <= q) && ((q - k) <= WIN);
}

#if USE_ASYNC
// the VDST register holds the LDS address; low 32 bits of a generic LDS pointer are the
// LDS byte offset (flat aperture rule), so pass that as the VDST operand.
__device__ __forceinline__ void async_ld16(void* lds, const void* gaddr) {
  uint32_t loff = (uint32_t)(uintptr_t)lds;
  asm volatile("global_load_async_to_lds_b128 %0, %1, off"
               :: "v"(loff), "v"(gaddr) : "memory");
}
#endif

// ---------------------------------------------------------------- cast f32->bf16
__global__ __launch_bounds__(256) void cast_f32_bf16(const float* __restrict__ in,
                                                     bf16_t* __restrict__ out, int n4) {
  int i = blockIdx.x * 256 + threadIdx.x;
  if (i >= n4) return;
  float4 f = reinterpret_cast<const float4*>(in)[i];
  v4bf o;
  o.x = (bf16_t)f.x; o.y = (bf16_t)f.y; o.z = (bf16_t)f.z; o.w = (bf16_t)f.w;
  reinterpret_cast<v4bf*>(out)[i] = o;
}

// ---------------------------------------------------------------- bf16 WMMA GEMM
// C[M,N] = A[M,K] * Bt[N,K]^T. Block tile 128x256, 8 waves, 64x64 per wave,
// BK=32, double-buffered async LDS staging.
template<typename CT>
__global__ __launch_bounds__(256) void gemm_bf16_wmma(
    const bf16_t* __restrict__ A, const bf16_t* __restrict__ Bt,
    CT* __restrict__ C, int M, int N, int K)
{
  constexpr int BK = 32, PITCH = 40; // +8 bf16 pad per row
  __shared__ __align__(16) bf16_t As[2][128 * PITCH];
  __shared__ __align__(16) bf16_t Bs[2][256 * PITCH];

  const int tid  = threadIdx.x;
  const int wave = tid >> 5;
  const int lane = tid & 31;
  const int half = lane >> 4;
  const int l16  = lane & 15;
  const int wm   = wave >> 2;           // 0..1 -> 64 rows
  const int wn   = wave & 3;            // 0..3 -> 64 cols
  const int bM   = blockIdx.y * 128;
  const int bN   = blockIdx.x * 256;

  v8f acc[4][4];
  #pragma unroll
  for (int i = 0; i < 4; ++i)
    #pragma unroll
    for (int j = 0; j < 4; ++j)
      #pragma unroll
      for (int e = 0; e < 8; ++e) acc[i][j][e] = 0.f;

  auto stage = [&](int b, int k0) {
    #pragma unroll
    for (int it = 0; it < 2; ++it) {          // A: 128 rows x 4 chunks
      int c = tid + it * 256;
      int r = c >> 2, cc = (c & 3) * 8;
#if USE_ASYNC
      async_ld16(As[b] + r * PITCH + cc, A + (size_t)(bM + r) * K + k0 + cc);
#else
      *(v8bf*)(As[b] + r * PITCH + cc) = *(const v8bf*)(A + (size_t)(bM + r) * K + k0 + cc);
#endif
    }
    #pragma unroll
    for (int it = 0; it < 4; ++it) {          // B: 256 rows x 4 chunks
      int c = tid + it * 256;
      int r = c >> 2, cc = (c & 3) * 8;
#if USE_ASYNC
      async_ld16(Bs[b] + r * PITCH + cc, Bt + (size_t)(bN + r) * K + k0 + cc);
#else
      *(v8bf*)(Bs[b] + r * PITCH + cc) = *(const v8bf*)(Bt + (size_t)(bN + r) * K + k0 + cc);
#endif
    }
  };

  stage(0, 0);                                 // prologue: tile 0 in flight
  int buf = 0;
  for (int k0 = 0; k0 < K; k0 += BK) {
#if USE_ASYNC
    if (k0 + BK < K) {
      stage(buf ^ 1, k0 + BK);                 // 6 new async ops in flight
      asm volatile("s_wait_asynccnt 0x6" ::: "memory");  // prev tile landed
    } else {
      asm volatile("s_wait_asynccnt 0x0" ::: "memory");
    }
#endif
    __syncthreads();

    v16bf af[4], bfr[4];
    #pragma unroll
    for (int mt = 0; mt < 4; ++mt) {           // A: lane=row, K 0-7/16-23 vs 8-15/24-31
      const bf16_t* p = As[buf] + (wm * 64 + mt * 16 + l16) * PITCH;
      af[mt] = cat16(*(const v8bf*)(p + half * 8), *(const v8bf*)(p + 16 + half * 8));
    }
    #pragma unroll
    for (int nt = 0; nt < 4; ++nt) {           // B: lane=col, K 0-7/16-23 vs 8-15/24-31
      const bf16_t* p = Bs[buf] + (wn * 64 + nt * 16 + l16) * PITCH + half * 8;
      bfr[nt] = cat16(*(const v8bf*)p, *(const v8bf*)(p + 16));
    }
    #pragma unroll
    for (int mt = 0; mt < 4; ++mt)
      #pragma unroll
      for (int nt = 0; nt < 4; ++nt)
        acc[mt][nt] = __builtin_amdgcn_wmma_f32_16x16x32_bf16(
            false, af[mt], false, bfr[nt], (short)0, acc[mt][nt], false, false);

    buf ^= 1;
    __syncthreads();
  }

  #pragma unroll
  for (int mt = 0; mt < 4; ++mt)
    #pragma unroll
    for (int nt = 0; nt < 4; ++nt)
      #pragma unroll
      for (int i = 0; i < 8; ++i) {
        int row = bM + wm * 64 + mt * 16 + half * 8 + i;
        int col = bN + wn * 64 + nt * 16 + l16;
        C[(size_t)row * N + col] = (CT)acc[mt ^ 0][nt][i];
      }
}

// ---------------------------------------------------------------- V -> Vt[e][t]
__global__ __launch_bounds__(256) void transpose_v(const bf16_t* __restrict__ qkv,
                                                   bf16_t* __restrict__ vt) {
  __shared__ bf16_t tile[32][33];
  int tx = threadIdx.x & 31, ty = threadIdx.x >> 5;     // 32 x 8
  int t0 = blockIdx.x * 32, e0 = blockIdx.y * 32;
  #pragma unroll
  for (int j = 0; j < 32; j += 8)
    tile[ty + j][tx] = qkv[(size_t)(t0 + ty + j) * EQKV + 2 * DMODEL + e0 + tx];
  __syncthreads();
  #pragma unroll
  for (int j = 0; j < 32; j += 8)
    vt[(size_t)(e0 + ty + j) * T_SEQ + t0 + tx] = tile[tx][ty + j];
}

// ---------------------------------------------------------------- flash attention
__global__ __launch_bounds__(128) void attn_wmma(
    const bf16_t* __restrict__ qkv,   // [T][3072]  (Q | K | V)
    const bf16_t* __restrict__ vt,    // [1024][T]
    bf16_t* __restrict__ ao)          // [T][1024]
{
  constexpr int KP = 72, VP = 40, PP = 40;
  __shared__ __align__(16) bf16_t Ks[32 * KP];       // [key][d]
  __shared__ __align__(16) bf16_t Vs[64 * VP];       // [d][key]
  __shared__ __align__(16) bf16_t Ps[4 * 16 * PP];   // per-wave P tile

  const int tid  = threadIdx.x;
  const int wave = tid >> 5;
  const int lane = tid & 31;
  const int half = lane >> 4;
  const int l16  = lane & 15;
  const int h    = blockIdx.y;
  const int q0   = blockIdx.x * 64;
  const int qw   = q0 + wave * 16;

  // Q fragments for this wave's 16 rows (d split 0..31 / 32..63)
  v16bf qf[2];
  {
    const bf16_t* qrow = qkv + (size_t)(qw + l16) * EQKV + h * DHEAD;
    #pragma unroll
    for (int s = 0; s < 2; ++s) {
      const bf16_t* p = qrow + s * 32 + half * 8;
      qf[s] = cat16(*(const v8bf*)p, *(const v8bf*)(p + 16));
    }
  }

  v8f o[4];
  float m_i[8], l_i[8];
  #pragma unroll
  for (int dt = 0; dt < 4; ++dt)
    #pragma unroll
    for (int e = 0; e < 8; ++e) o[dt][e] = 0.f;
  #pragma unroll
  for (int i = 0; i < 8; ++i) { m_i[i] = -1e30f; l_i[i] = 0.f; }

  bf16_t* pw = Ps + wave * 16 * PP;

  int kb = 32;
  { int lo = q0 - WIN; if (lo > 32) kb = (lo >> 5) << 5; }

  for (int k0 = 0; k0 <= q0 + 63; k0 = (k0 == 0) ? kb : k0 + 32) {
    // cooperative staging: K tile 32x64, Vt tile 64x32
    #pragma unroll
    for (int it = 0; it < 2; ++it) {
      int c = tid + it * 128;
      { int r = c >> 3, cc = (c & 7) * 8;
        *(v8bf*)(Ks + r * KP + cc) =
          *(const v8bf*)(qkv + (size_t)(k0 + r) * EQKV + DMODEL + h * DHEAD + cc); }
      { int r = c >> 2, cc = (c & 3) * 8;
        *(v8bf*)(Vs + r * VP + cc) =
          *(const v8bf*)(vt + (size_t)(h * DHEAD + r) * T_SEQ + k0 + cc); }
    }
    __syncthreads();

    bool active = (k0 == 0) || ((k0 <= qw + 15) && (k0 + 31 >= qw - WIN));
    if (active) {
      // ---- scores: S0 = Q*K[k0..k0+15]^T, S1 = Q*K[k0+16..k0+31]^T
      v8f s0, s1;
      #pragma unroll
      for (int e = 0; e < 8; ++e) { s0[e] = 0.f; s1[e] = 0.f; }
      #pragma unroll
      for (int ds = 0; ds < 2; ++ds) {
        const bf16_t* p0 = Ks + (l16)      * KP + ds * 32 + half * 8;
        const bf16_t* p1 = Ks + (16 + l16) * KP + ds * 32 + half * 8;
        v16bf b0 = cat16(*(const v8bf*)p0, *(const v8bf*)(p0 + 16));
        v16bf b1 = cat16(*(const v8bf*)p1, *(const v8bf*)(p1 + 16));
        s0 = __builtin_amdgcn_wmma_f32_16x16x32_bf16(false, qf[ds], false, b0, (short)0, s0, false, false);
        s1 = __builtin_amdgcn_wmma_f32_16x16x32_bf16(false, qf[ds], false, b1, (short)0, s1, false, false);
      }

      // ---- online softmax (rows = vgpr+8*half, cols = lanes)
      const float scale = 0.125f; // 1/sqrt(64)
      #pragma unroll
      for (int i = 0; i < 8; ++i) {
        int q  = qw + half * 8 + i;
        float v0 = mask_ok(q, k0 + l16)      ? s0[i] * scale : -1e30f;
        float v1 = mask_ok(q, k0 + 16 + l16) ? s1[i] * scale : -1e30f;
        float t = fmaxf(v0, v1);
        #pragma unroll
        for (int off = 1; off < 16; off <<= 1) t = fmaxf(t, __shfl_xor(t, off, 32));
        float mn = fmaxf(m_i[i], t);
        float alpha = __expf(m_i[i] - mn);
        float e0 = __expf(v0 - mn), e1 = __expf(v1 - mn);
        float rs = e0 + e1;
        #pragma unroll
        for (int off = 1; off < 16; off <<= 1) rs += __shfl_xor(rs, off, 32);
        l_i[i] = l_i[i] * alpha + rs;
        m_i[i] = mn;
        #pragma unroll
        for (int dt = 0; dt < 4; ++dt) o[dt][i] *= alpha;
        pw[(half * 8 + i) * PP + l16]      = (bf16_t)e0;
        pw[(half * 8 + i) * PP + 16 + l16] = (bf16_t)e1;
      }

      // ---- P (C-layout in LDS) -> A-fragment, then O += P * V
      v16bf pf;
      { const bf16_t* p = pw + l16 * PP;
        pf = cat16(*(const v8bf*)(p + half * 8), *(const v8bf*)(p + 16 + half * 8)); }
      #pragma unroll
      for (int dt = 0; dt < 4; ++dt) {
        const bf16_t* p = Vs + (dt * 16 + l16) * VP + half * 8;
        v16bf bv = cat16(*(const v8bf*)p, *(const v8bf*)(p + 16));
        o[dt] = __builtin_amdgcn_wmma_f32_16x16x32_bf16(false, pf, false, bv, (short)0, o[dt], false, false);
      }
    }
    __syncthreads();
  }

  // ---- normalize and write
  #pragma unroll
  for (int dt = 0; dt < 4; ++dt)
    #pragma unroll
    for (int i = 0; i < 8; ++i) {
      int row = qw + half * 8 + i;
      float val = o[dt][i] / l_i[i];
      ao[(size_t)row * DMODEL + h * DHEAD + dt * 16 + l16] = (bf16_t)val;
    }
}

// ---------------------------------------------------------------- launch
extern "C" void kernel_launch(void* const* d_in, const int* in_sizes, int n_in,
                              void* d_out, int out_size, void* d_ws, size_t ws_size,
                              hipStream_t stream) {
  (void)in_sizes; (void)n_in; (void)out_size; (void)ws_size;
  const float* x    = (const float*)d_in[0];
  const float* wqkv = (const float*)d_in[1];
  const float* wout = (const float*)d_in[2];
  float* out = (float*)d_out;

  bf16_t* xb    = (bf16_t*)d_ws;
  bf16_t* wqkvb = xb    + (size_t)T_SEQ * DMODEL;
  bf16_t* woutb = wqkvb + (size_t)EQKV  * DMODEL;
  bf16_t* qkvb  = woutb + (size_t)DMODEL * DMODEL;
  bf16_t* vtb   = qkvb  + (size_t)T_SEQ * EQKV;
  bf16_t* aob   = vtb   + (size_t)DMODEL * T_SEQ;

  cast_f32_bf16<<<(T_SEQ * DMODEL / 4 + 255) / 256, 256, 0, stream>>>(x, xb, T_SEQ * DMODEL / 4);
  cast_f32_bf16<<<(EQKV * DMODEL / 4 + 255) / 256, 256, 0, stream>>>(wqkv, wqkvb, EQKV * DMODEL / 4);
  cast_f32_bf16<<<(DMODEL * DMODEL / 4 + 255) / 256, 256, 0, stream>>>(wout, woutb, DMODEL * DMODEL / 4);

  gemm_bf16_wmma<bf16_t><<<dim3(EQKV / 256, T_SEQ / 128), 256, 0, stream>>>(
      xb, wqkvb, qkvb, T_SEQ, EQKV, DMODEL);

  transpose_v<<<dim3(T_SEQ / 32, DMODEL / 32), 256, 0, stream>>>(qkvb, vtb);

  attn_wmma<<<dim3(T_SEQ / 64, NHEAD), 128, 0, stream>>>(qkvb, vtb, aob);

  gemm_bf16_wmma<float><<<dim3(DMODEL / 256, T_SEQ / 128), 256, 0, stream>>>(
      aob, woutb, out, T_SEQ, DMODEL, DMODEL);
}
